// Generator_62070867362235
// MI455X (gfx1250) — matrix operand order repaired and checked by reading, not compiled, not run on hardware
//
#include <hip/hip_runtime.h>

// ---------------------------------------------------------------------------
// CDNA5 (gfx1250) WMMA transformer generator, v2.
//  - All weights pre-converted to f16 and pre-swizzled into WMMA B-fragment
//    order (one contiguous 32B load per lane per k-step, no cvt in GEMM loop).
//  - Residual stream kept fp32; f16 mirrors maintained for all GEMM A inputs.
//  - GEMMs / attention use v_wmma_f32_16x16x32_f16 with fp32 accumulation.
// ---------------------------------------------------------------------------

typedef __attribute__((ext_vector_type(16))) _Float16 v16h;
typedef __attribute__((ext_vector_type(8)))  float    v8f;

union HFrag { v16h v; _Float16 h[16]; };

// A-matrix fragment (16x32 f16) from f16 row-major source, row stride lda.
// Layout: lane -> M = m0 + (lane&15); half i -> K = kb + (lane>>4)*16 + i
// Per lane this is 16 contiguous halfs (matches the contiguous-k B fragments).
__device__ inline v16h load_a_frag_f16(const _Float16* __restrict__ A, int lda, int m0, int kb) {
  int lane = threadIdx.x & 31;
  int lo = lane & 15, hi = lane >> 4;
  const _Float16* row = A + (size_t)(m0 + lo) * lda;
  HFrag f;
#pragma unroll
  for (int i = 0; i < 16; ++i) {
    int k = kb + (hi << 4) + i;
    f.h[i] = row[k];
  }
  return f.v;
}

// B fragment where logical B[k][n] = src[n*ldw + k] (f16, N x K row-major).
// Per lane: 16 contiguous halfs (32B, aligned) -> single vector load.
__device__ inline v16h load_bT_frag_f16(const _Float16* __restrict__ src, int ldw, int n0, int kb) {
  int lane = threadIdx.x & 31;
  int lo = lane & 15, hi = lane >> 4;
  return *(const v16h*)(src + (size_t)(n0 + lo) * ldw + kb + (hi << 4));
}

// B fragment from f16 row-major K x N source (strided; used only for 32x64 V tiles).
__device__ inline v16h load_b_frag_f16(const _Float16* __restrict__ W, int ldw, int n0, int kb) {
  int lane = threadIdx.x & 31;
  int lo = lane & 15, hi = lane >> 4;
  HFrag f;
#pragma unroll
  for (int i = 0; i < 16; ++i) {
    int k = kb + (hi << 4) + i;
    f.h[i] = W[(size_t)k * ldw + (n0 + lo)];
  }
  return f.v;
}

// ---------------------------------------------------------------------------
// Weight pre-swizzle: fp32 K x N row-major -> f16 fragment blocks.
// Block (ntile, kbi) holds a 32x16 B-fragment: lane l's 16 halfs contiguous at
// out[blk*512 + l*16]; element i = (f16) W[(kb + (l>>4)*16 + i)*N + ntile*16 + (l&15)]
// Blocks ordered kbi-fastest within an ntile -> GEMM K-loop streams contiguously.
// ---------------------------------------------------------------------------
__global__ __launch_bounds__(256)
void swizzle_w_kernel(const float* __restrict__ W, _Float16* __restrict__ out, int N, int K)
{
  int gid = blockIdx.x * blockDim.x + threadIdx.x;
  int kblocks = K >> 5;
  int total = (N >> 4) * kblocks * 32;
  if (gid >= total) return;
  int lane = gid & 31;
  int blk  = gid >> 5;
  int ntile = blk / kblocks;
  int kbi   = blk - ntile * kblocks;
  int n  = (ntile << 4) + (lane & 15);
  int kb = (kbi << 5) + ((lane >> 4) << 4);
  _Float16* o = out + (size_t)blk * 512 + lane * 16;
  const float* src = W + (size_t)kb * N + n;
#pragma unroll
  for (int i = 0; i < 16; ++i) o[i] = (_Float16)src[(size_t)i * N];
}

// fp32 -> f16 flat cast (for the `noise` input).
__global__ __launch_bounds__(256)
void cast_f16_kernel(const float* __restrict__ in, _Float16* __restrict__ out, int total)
{
  int idx = blockIdx.x * blockDim.x + threadIdx.x;
  if (idx < total) out[idx] = (_Float16)in[idx];
}

// ---------------------------------------------------------------------------
// GEMM: C[m,n] = act( sum_k A16[m,k] * Wsw[k,n] + bias[n] )
// OUTM bit0: write fp32 C32; bit1: write f16 C16. One wave = one 16x16 tile.
// Weights are L2-resident f16 fragments: one 32B contiguous load per k-step.
// ---------------------------------------------------------------------------
template <int ACT, int OUTM>
__global__ __launch_bounds__(128)
void gemm_wmma_kernel(const _Float16* __restrict__ A, int sA,
                      const _Float16* __restrict__ Wsw,
                      const float* __restrict__ bias,
                      float* __restrict__ C32, _Float16* __restrict__ C16,
                      int sC, int N, int K)
{
  int wave = threadIdx.x >> 5;
  int lane = threadIdx.x & 31;
  int lo = lane & 15, hi = lane >> 4;
  int ntiles = N >> 4;
  int ntile = blockIdx.x * (blockDim.x >> 5) + wave;
  if (ntile >= ntiles) return;            // wave-uniform; EXEC all-1 for WMMA
  int n0 = ntile << 4;
  int m0 = blockIdx.y << 4;
  int kblocks = K >> 5;

  const _Float16* wp = Wsw + ((size_t)ntile * kblocks) * 512 + lane * 16;
  v8f acc = {};
  for (int kbi = 0; kbi < kblocks; ++kbi) {
    v16h a = load_a_frag_f16(A, sA, m0, kbi << 5);
    v16h b = *(const v16h*)wp;            // contiguous 32B per lane
    wp += 512;
    acc = __builtin_amdgcn_wmma_f32_16x16x32_f16(false, a, false, b, (short)0, acc, false, false);
  }

  int n = n0 + lo;
  float bv = bias ? bias[n] : 0.0f;
#pragma unroll
  for (int v = 0; v < 8; ++v) {
    int m = m0 + v + (hi << 3);           // C/D layout: lanes 16-31 hold M+8
    float val = acc[v] + bv;
    if (ACT == 1) val = fmaxf(val, 0.0f);
    if (OUTM & 1) C32[(size_t)m * sC + n] = val;
    if (OUTM & 2) C16[(size_t)m * sC + n] = (_Float16)val;
  }
}

// ---------------------------------------------------------------------------
// Attention core: one wave per (b,h). S=32, D=64, all-f16 operands.
// scores = Q K^T / 8, mask keys > t, softmax rows, O = P V (f16 out).
// ---------------------------------------------------------------------------
__global__ __launch_bounds__(32)
void mha_wmma_kernel(const _Float16* __restrict__ qh,
                     const _Float16* __restrict__ kh,
                     const _Float16* __restrict__ vh,
                     _Float16* __restrict__ o,
                     int t)
{
  const int S = 32, E = 512, D = 64;
  int b = blockIdx.x >> 3;
  int h = blockIdx.x & 7;
  int lane = threadIdx.x & 31;
  int lo = lane & 15, hi = lane >> 4;

  const _Float16* qb = qh + (size_t)b * S * E + h * D;
  const _Float16* kb = kh + (size_t)b * S * E + h * D;
  const _Float16* vb = vh + (size_t)b * S * E + h * D;

  // ---- scores: 2x2 tiles of 16x16, K = D = 64 ----
  v8f sc[2][2];
#pragma unroll
  for (int ti = 0; ti < 2; ++ti) {
#pragma unroll
    for (int tj = 0; tj < 2; ++tj) {
      v8f acc = {};
#pragma unroll
      for (int kk = 0; kk < D; kk += 32) {
        v16h a  = load_a_frag_f16(qb, E, ti * 16, kk);
        v16h bf = load_bT_frag_f16(kb, E, tj * 16, kk);  // B[k][n] = K[n][k]
        acc = __builtin_amdgcn_wmma_f32_16x16x32_f16(false, a, false, bf, (short)0, acc, false, false);
      }
      sc[ti][tj] = acc;
    }
  }

  // ---- scale + key mask (valid: key <= t) ----
  const float scale = 0.125f;             // 1/sqrt(64)
#pragma unroll
  for (int ti = 0; ti < 2; ++ti)
#pragma unroll
    for (int tj = 0; tj < 2; ++tj) {
      bool valid = ((tj * 16 + lo) <= t);
#pragma unroll
      for (int v = 0; v < 8; ++v) {
        float s = sc[ti][tj][v] * scale;
        sc[ti][tj][v] = valid ? s : -1.0e9f;
      }
    }

  // ---- masked softmax per row (rows in (vgpr, lane-half); cols across 16 lanes) ----
  __shared__ _Float16 pl[32 * 32];
#pragma unroll
  for (int ti = 0; ti < 2; ++ti) {
#pragma unroll
    for (int v = 0; v < 8; ++v) {
      float a0 = sc[ti][0][v], a1 = sc[ti][1][v];
      float mx = fmaxf(a0, a1);
#pragma unroll
      for (int off = 1; off < 16; off <<= 1) mx = fmaxf(mx, __shfl_xor(mx, off, 32));
      float e0 = __expf(a0 - mx), e1 = __expf(a1 - mx);
      float sm = e0 + e1;
#pragma unroll
      for (int off = 1; off < 16; off <<= 1) sm += __shfl_xor(sm, off, 32);
      float inv = 1.0f / sm;
      int m = ti * 16 + v + (hi << 3);
      pl[m * 32 + lo]      = (_Float16)(e0 * inv);
      pl[m * 32 + 16 + lo] = (_Float16)(e1 * inv);
    }
  }
  __syncthreads();

  // ---- O = P (32x32) @ V (32x64): 2x4 tiles, K=32 ----
  _Float16* ob = o + (size_t)b * S * E + h * D;
#pragma unroll
  for (int ti = 0; ti < 2; ++ti) {
    v16h a = load_a_frag_f16(pl, 32, ti * 16, 0);
#pragma unroll
    for (int tc = 0; tc < 4; ++tc) {
      v16h bf = load_b_frag_f16(vb, E, tc * 16, 0);
      v8f acc = {};
      acc = __builtin_amdgcn_wmma_f32_16x16x32_f16(false, a, false, bf, (short)0, acc, false, false);
      int n = tc * 16 + lo;
#pragma unroll
      for (int v = 0; v < 8; ++v)
        ob[(size_t)(ti * 16 + v + (hi << 3)) * E + n] = (_Float16)acc[v];
    }
  }
}

// ---------------------------------------------------------------------------
// y = LayerNorm(x + r) * g + b over E=512.  One wave per row.
// Writes fp32 residual stream + f16 mirror for the next GEMM.
// ---------------------------------------------------------------------------
__global__ __launch_bounds__(256)
void add_ln_kernel(const float* __restrict__ x, const float* __restrict__ r,
                   const float* __restrict__ g, const float* __restrict__ bb,
                   float* __restrict__ out32, _Float16* __restrict__ out16, int rows)
{
  const int E = 512;
  int wave = threadIdx.x >> 5, lane = threadIdx.x & 31;
  int row = blockIdx.x * (blockDim.x >> 5) + wave;
  if (row >= rows) return;
  const float* xr = x + (size_t)row * E;
  const float* rr = r + (size_t)row * E;
  float vals[16];
  float s = 0.0f, s2 = 0.0f;
#pragma unroll
  for (int i = 0; i < 16; ++i) {
    float v = xr[lane + i * 32] + rr[lane + i * 32];
    vals[i] = v; s += v; s2 += v * v;
  }
#pragma unroll
  for (int off = 1; off < 32; off <<= 1) { s += __shfl_xor(s, off, 32); s2 += __shfl_xor(s2, off, 32); }
  float mean = s * (1.0f / 512.0f);
  float var  = s2 * (1.0f / 512.0f) - mean * mean;
  float inv  = rsqrtf(var + 1e-5f);
  float*    o32 = out32 + (size_t)row * E;
  _Float16* o16 = out16 + (size_t)row * E;
#pragma unroll
  for (int i = 0; i < 16; ++i) {
    int e = lane + i * 32;
    float y = (vals[i] - mean) * inv * g[e] + bb[e];
    o32[e] = y;
    o16[e] = (_Float16)y;
  }
}

// x = Ybuf + positional encoding; writes fp32 + f16 mirror.
__global__ __launch_bounds__(256)
void add_pe_kernel(const float* __restrict__ Y, float* __restrict__ x32,
                   _Float16* __restrict__ x16, int total)
{
  int idx = blockIdx.x * blockDim.x + threadIdx.x;
  if (idx >= total) return;
  const int E = 512, S = 32;
  int e = idx & (E - 1);
  int s = (idx >> 9) & (S - 1);
  float i2  = (float)(e & ~1);
  float ang = (float)s * __powf(10000.0f, -i2 / (float)E);
  float pe  = (e & 1) ? __cosf(ang) : __sinf(ang);
  float v = Y[idx] + pe;
  x32[idx] = v;
  x16[idx] = (_Float16)v;
}

// ---------------------------------------------------------------------------
// Head: per batch b -> softmax over V=30000 into out[b,t,:];
// argmax (first occurrence) -> Ybuf[b,t,:] = w2v[argmax].
// ---------------------------------------------------------------------------
__global__ __launch_bounds__(256)
void head_kernel(const float* __restrict__ logits,  // B x V
                 const float* __restrict__ w2v,     // V x E
                 float* __restrict__ out,           // B x S x V
                 float* __restrict__ Ybuf,          // B x S x E
                 int t)
{
  const int V = 30000, E = 512, S = 32;
  int b = blockIdx.x;
  int tid = threadIdx.x;
  __shared__ float smax[256];
  __shared__ int   sidx[256];
  __shared__ float ssum[256];
  const float* lg = logits + (size_t)b * V;

  float mx = -3.4e38f; int mi = 0;
  for (int v = tid; v < V; v += 256) { float val = lg[v]; if (val > mx) { mx = val; mi = v; } }
  smax[tid] = mx; sidx[tid] = mi;
  __syncthreads();
  for (int st = 128; st > 0; st >>= 1) {
    if (tid < st) {
      float om = smax[tid + st]; int oi = sidx[tid + st];
      if (om > smax[tid] || (om == smax[tid] && oi < sidx[tid])) { smax[tid] = om; sidx[tid] = oi; }
    }
    __syncthreads();
  }
  float gmax = smax[0];
  int   gidx = sidx[0];

  float s = 0.0f;
  for (int v = tid; v < V; v += 256) s += __expf(lg[v] - gmax);
  ssum[tid] = s;
  __syncthreads();
  for (int st = 128; st > 0; st >>= 1) { if (tid < st) ssum[tid] += ssum[tid + st]; __syncthreads(); }
  float inv = 1.0f / ssum[0];

  float* ob = out + ((size_t)b * S + t) * V;
  for (int v = tid; v < V; v += 256) ob[v] = __expf(lg[v] - gmax) * inv;

  const float* wv = w2v + (size_t)gidx * E;
  float* yb = Ybuf + ((size_t)b * S + t) * E;
  for (int e = tid; e < E; e += 256) yb[e] = wv[e];
}

// ---------------------------------------------------------------------------
// Host driver
// ---------------------------------------------------------------------------
extern "C" void kernel_launch(void* const* d_in, const int* in_sizes, int n_in,
                              void* d_out, int out_size, void* d_ws, size_t ws_size,
                              hipStream_t stream)
{
  (void)in_sizes; (void)n_in; (void)out_size; (void)ws_size;
  const int B = 32, S = 32, E = 512, HF = 2048, V = 30000, NB = 2, NO = 1;

  const float* noise    = (const float*)d_in[0];
  const float* seeds    = (const float*)d_in[1];
  const float* inemb_W  = (const float*)d_in[2];
  const float* inemb_b  = (const float*)d_in[3];
  const float* gaW      = (const float*)d_in[4];
  const float* gab      = (const float*)d_in[5];
  const float* gcW      = (const float*)d_in[6];
  const float* gcb      = (const float*)d_in[7];
  const float* gF1      = (const float*)d_in[8];
  const float* gb1      = (const float*)d_in[9];
  const float* gF2      = (const float*)d_in[10];
  const float* gb2      = (const float*)d_in[11];
  const float* glng     = (const float*)d_in[12];
  const float* glnb     = (const float*)d_in[13];
  const float* oaW      = (const float*)d_in[14];
  const float* oab      = (const float*)d_in[15];
  const float* oF1      = (const float*)d_in[16];
  const float* ob1      = (const float*)d_in[17];
  const float* oF2      = (const float*)d_in[18];
  const float* ob2      = (const float*)d_in[19];
  const float* olng     = (const float*)d_in[20];
  const float* olnb     = (const float*)d_in[21];
  const float* w2v      = (const float*)d_in[22];
  const float* soft_W   = (const float*)d_in[23];
  const float* soft_b   = (const float*)d_in[24];
  float* out = (float*)d_out;

  // ---- workspace layout ----
  char* ws = (char*)d_ws;
  size_t off = 0;
  auto allocB = [&](size_t bytes) -> void* {
    void* p = ws + off;
    off += (bytes + 255) & ~(size_t)255;
    return p;
  };
  auto allocF = [&](size_t elems) -> float*     { return (float*)allocB(elems * 4); };
  auto allocH = [&](size_t elems) -> _Float16*  { return (_Float16*)allocB(elems * 2); };

  const size_t BSE = (size_t)B * S * E;
  // activations
  float*    Ybuf   = allocF(BSE);
  float*    x32    = allocF(BSE);
  _Float16* x16    = allocH(BSE);
  float*    prj    = allocF(BSE);
  _Float16* w16    = allocH(BSE);
  _Float16* qh16   = allocH(BSE);
  _Float16* kh16   = allocH(BSE);
  _Float16* vh16   = allocH(BSE);
  _Float16* att16  = allocH(BSE);
  _Float16* tmp16  = allocH(BSE);
  _Float16* noise16= allocH(BSE);
  _Float16* ffn16  = allocH((size_t)B * S * HF);
  float*    logits = allocF((size_t)B * V);
  // swizzled f16 weights
  _Float16* sw_in[2];   for (int i = 0; i < 2; ++i)  sw_in[i]  = allocH((size_t)E * E);
  _Float16* sw_ga[2][4];for (int i = 0; i < NB; ++i) for (int j = 0; j < 4; ++j) sw_ga[i][j] = allocH((size_t)E * E);
  _Float16* sw_gc[2][4];for (int i = 0; i < NB; ++i) for (int j = 0; j < 4; ++j) sw_gc[i][j] = allocH((size_t)E * E);
  _Float16* sw_oa[1][4];for (int i = 0; i < NO; ++i) for (int j = 0; j < 4; ++j) sw_oa[i][j] = allocH((size_t)E * E);
  _Float16* sw_gF1[2];  for (int i = 0; i < NB; ++i) sw_gF1[i] = allocH((size_t)E * HF);
  _Float16* sw_gF2[2];  for (int i = 0; i < NB; ++i) sw_gF2[i] = allocH((size_t)HF * E);
  _Float16* sw_oF1[1];  for (int i = 0; i < NO; ++i) sw_oF1[i] = allocH((size_t)E * HF);
  _Float16* sw_oF2[1];  for (int i = 0; i < NO; ++i) sw_oF2[i] = allocH((size_t)HF * E);
  _Float16* sw_soft = allocH((size_t)E * V);

  auto swz = [&](const float* W, _Float16* o, int N, int K) {
    int total = (N >> 4) * (K >> 5) * 32;
    swizzle_w_kernel<<<(total + 255) / 256, 256, 0, stream>>>(W, o, N, K);
  };

  // ---- one-time per call: weight swizzle + input casts ----
  for (int i = 0; i < 2; ++i)  swz(inemb_W + (size_t)i * E * E, sw_in[i], E, E);
  for (int i = 0; i < NB; ++i) for (int j = 0; j < 4; ++j) {
    swz(gaW + ((size_t)i * 4 + j) * E * E, sw_ga[i][j], E, E);
    swz(gcW + ((size_t)i * 4 + j) * E * E, sw_gc[i][j], E, E);
  }
  for (int i = 0; i < NO; ++i) for (int j = 0; j < 4; ++j)
    swz(oaW + ((size_t)i * 4 + j) * E * E, sw_oa[i][j], E, E);
  for (int i = 0; i < NB; ++i) {
    swz(gF1 + (size_t)i * E * HF, sw_gF1[i], HF, E);
    swz(gF2 + (size_t)i * HF * E, sw_gF2[i], E, HF);
  }
  for (int i = 0; i < NO; ++i) {
    swz(oF1 + (size_t)i * E * HF, sw_oF1[i], HF, E);
    swz(oF2 + (size_t)i * HF * E, sw_oF2[i], E, HF);
  }
  swz(soft_W, sw_soft, V, E);
  cast_f16_kernel<<<(int)(BSE / 256), 256, 0, stream>>>(noise, noise16, (int)BSE);
  hipMemcpyAsync(Ybuf, seeds, BSE * 4, hipMemcpyDeviceToDevice, stream);

  auto gemm = [&](const _Float16* A, int sA, const _Float16* Wsw, const float* bias,
                  float* C32, _Float16* C16, int sC, int M, int N, int K, int act, int outm) {
    const int WPB = 4;
    dim3 grid((N / 16 + WPB - 1) / WPB, M / 16);
    dim3 block(32 * WPB);
    if (act == 1)
      gemm_wmma_kernel<1, 2><<<grid, block, 0, stream>>>(A, sA, Wsw, bias, C32, C16, sC, N, K);
    else if (outm == 1)
      gemm_wmma_kernel<0, 1><<<grid, block, 0, stream>>>(A, sA, Wsw, bias, C32, C16, sC, N, K);
    else
      gemm_wmma_kernel<0, 2><<<grid, block, 0, stream>>>(A, sA, Wsw, bias, C32, C16, sC, N, K);
  };

  // ---- w = (noise @ W0 + b0) @ W1 + b1 (f16-only: consumed only as GEMM input) ----
  gemm(noise16, E, sw_in[0], inemb_b,     nullptr, tmp16, E, B * S, E, E, 0, 2);
  gemm(tmp16,   E, sw_in[1], inemb_b + E, nullptr, w16,   E, B * S, E, E, 0, 2);

  // ---- sequential decode ----
  for (int t = 0; t < S; ++t) {
    add_pe_kernel<<<(int)(BSE / 256), 256, 0, stream>>>(Ybuf, x32, x16, (int)BSE);

    auto mha_block = [&](const _Float16* kvin, _Float16* const* Wp4, const float* bp,
                         const float* lg, const float* lb) {
      gemm(x16,  E, Wp4[0], bp + 0 * E, nullptr, qh16, E, B * S, E, E, 0, 2);
      gemm(kvin, E, Wp4[1], bp + 1 * E, nullptr, kh16, E, B * S, E, E, 0, 2);
      gemm(kvin, E, Wp4[2], bp + 2 * E, nullptr, vh16, E, B * S, E, E, 0, 2);
      mha_wmma_kernel<<<B * 8, 32, 0, stream>>>(qh16, kh16, vh16, att16, t);
      gemm(att16, E, Wp4[3], bp + 3 * E, prj, nullptr, E, B * S, E, E, 0, 1);
      add_ln_kernel<<<(B * S) / 8, 256, 0, stream>>>(x32, prj, lg, lb, x32, x16, B * S);
    };
    auto ffn_block = [&](const _Float16* W1s, const float* b1, const _Float16* W2s, const float* b2,
                         const float* lg, const float* lb) {
      gemm(x16,   E,  W1s, b1, nullptr, ffn16, HF, B * S, HF, E, 1, 2);
      gemm(ffn16, HF, W2s, b2, prj, nullptr, E,  B * S, E, HF, 0, 1);
      add_ln_kernel<<<(B * S) / 8, 256, 0, stream>>>(x32, prj, lg, lb, x32, x16, B * S);
    };

    for (int i = 0; i < NB; ++i) {
      const size_t bB = (size_t)i * 4 * E;
      mha_block(x16, sw_ga[i], gab + bB, glng + ((size_t)i * 3 + 0) * E, glnb + ((size_t)i * 3 + 0) * E);
      mha_block(w16, sw_gc[i], gcb + bB, glng + ((size_t)i * 3 + 1) * E, glnb + ((size_t)i * 3 + 1) * E);
      ffn_block(sw_gF1[i], gb1 + (size_t)i * HF, sw_gF2[i], gb2 + (size_t)i * E,
                glng + ((size_t)i * 3 + 2) * E, glnb + ((size_t)i * 3 + 2) * E);
    }
    for (int j = 0; j < NO; ++j) {
      const size_t bB = (size_t)j * 4 * E;
      mha_block(x16, sw_oa[j], oab + bB, olng + ((size_t)j * 2 + 0) * E, olnb + ((size_t)j * 2 + 0) * E);
      ffn_block(sw_oF1[j], ob1 + (size_t)j * HF, sw_oF2[j], ob2 + (size_t)j * E,
                olng + ((size_t)j * 2 + 1) * E, olnb + ((size_t)j * 2 + 1) * E);
    }

    // logits = x[:, t, :] @ soft_W + soft_b  (strided token gather over the f16 mirror)
    gemm(x16 + (size_t)t * E, S * E, sw_soft, soft_b, logits, nullptr, V, B, V, E, 0, 1);
    head_kernel<<<B, 256, 0, stream>>>(logits, w2v, out, Ybuf, t);
  }
}
